// PokerEq3Net_1047972021004
// MI455X (gfx1250) — matrix operand order repaired and checked by reading, not compiled, not run on hardware
//
#include <hip/hip_runtime.h>
#include <hip/hip_bf16.h>
#include <stdint.h>

// ---------------------------------------------------------------------------
// Types for CDNA5 WMMA (wave32).
// ---------------------------------------------------------------------------
typedef __attribute__((ext_vector_type(16))) __bf16 v16bf;
typedef __attribute__((ext_vector_type(8)))  __bf16 v8bf;
typedef __attribute__((ext_vector_type(8)))  float  v8f;

#define NB   64          // batch
#define NN   13          // set size n
#define ND   128         // channels D
#define NP   169         // n*n
#define NC   2197        // n*n*n
#define NOUT 10

__device__ __forceinline__ v16bf pack16(v8bf lo, v8bf hi) {
    v16bf r;
#pragma unroll
    for (int t = 0; t < 8; ++t) { r[t] = lo[t]; r[t + 8] = hi[t]; }
    return r;
}

__device__ __forceinline__ float frelu(float x) { return x > 0.0f ? x : 0.0f; }

// ---------------------------------------------------------------------------
// Encoder MLP: (suit,num) -> 128, per (b,i) row.  grid(13,64) x 128
// ---------------------------------------------------------------------------
__global__ __launch_bounds__(128) void encoder_kernel(
    const int* __restrict__ suits, const int* __restrict__ nums,
    const float* __restrict__ W1, const float* __restrict__ b1,
    const float* __restrict__ W2, const float* __restrict__ b2,
    const float* __restrict__ W3, const float* __restrict__ b3,
    float* __restrict__ x)
{
    __shared__ float h1[ND], h2[ND];
    const int e = threadIdx.x;
    const int i = blockIdx.x;
    const int b = blockIdx.y;
    const float fs = (float)suits[b * NN + i];
    const float fn = (float)nums [b * NN + i];

    h1[e] = frelu(fs * W1[e] + fn * W1[ND + e] + b1[e]);
    __syncthreads();
    float a = b2[e];
    for (int d = 0; d < ND; ++d) a += h1[d] * W2[d * ND + e];
    h2[e] = frelu(a);
    __syncthreads();
    float o = b3[e];
    for (int d = 0; d < ND; ++d) o += h2[d] * W3[d * ND + e];
    x[((size_t)(b * NN + i)) * ND + e] = o;
}

// ---------------------------------------------------------------------------
// T0[b, (i,j,k), d] = bf16( x[i,d]*x[j,d]*x[k,d] ).  grid(13,64) x 256
// ---------------------------------------------------------------------------
__global__ __launch_bounds__(256) void outer3_kernel(
    const float* __restrict__ x, __bf16* __restrict__ T)
{
    __shared__ float xl[NN * ND];
    const int tid = threadIdx.x;
    const int i   = blockIdx.x;
    const int b   = blockIdx.y;
    for (int idx = tid; idx < NN * ND; idx += 256)
        xl[idx] = x[(size_t)b * NN * ND + idx];
    __syncthreads();
    const float* xi = &xl[i * ND];
    for (int idx = tid; idx < NP * ND; idx += 256) {
        const int jk = idx >> 7;
        const int d  = idx & 127;
        const int j  = jk / NN;
        const int k  = jk - j * NN;
        const float v = xi[d] * xl[j * ND + d] * xl[k * ND + d];
        T[((size_t)b * NC + (size_t)i * NP + jk) * ND + d] = (__bf16)v;
    }
}

// ---------------------------------------------------------------------------
// Level-1 pools: Pi(j,k)=mean_i, Pj(i,k)=mean_j, Pk(i,j)=mean_k.
// grid(169,64) x 128 (thread = channel d, coalesced bf16 reads)
// ---------------------------------------------------------------------------
__global__ __launch_bounds__(128) void pool1_kernel(
    const __bf16* __restrict__ T,
    float* __restrict__ Pi, float* __restrict__ Pj, float* __restrict__ Pk)
{
    const int d    = threadIdx.x;
    const int pair = blockIdx.x;
    const int b    = blockIdx.y;
    const int hi   = pair / NN;
    const int lo   = pair - hi * NN;
    const size_t tb = (size_t)b * NC;
    float s1 = 0.f, s2 = 0.f, s3 = 0.f;
    for (int q = 0; q < NN; ++q) {
        s1 += (float)T[(tb + (size_t)q  * NP + pair)               * ND + d]; // over i; pair=(j,k)
        s2 += (float)T[(tb + (size_t)hi * NP + q * NN + lo)        * ND + d]; // over j; pair=(i,k)
        s3 += (float)T[(tb + (size_t)pair * NN + q)                * ND + d]; // over k; pair=(i,j)
    }
    const size_t ob = ((size_t)b * NP + pair) * ND + d;
    const float inv = 1.0f / (float)NN;
    Pi[ob] = s1 * inv; Pj[ob] = s2 * inv; Pk[ob] = s3 * inv;
}

// ---------------------------------------------------------------------------
// Level-2 pools: Pij(k)=mean_j Pi, Pik(j)=mean_k Pi, Pjk(i)=mean_k Pj.
// grid(13,64) x 128
// ---------------------------------------------------------------------------
__global__ __launch_bounds__(128) void pool2_kernel(
    const float* __restrict__ Pi, const float* __restrict__ Pj,
    float* __restrict__ Pij, float* __restrict__ Pik, float* __restrict__ Pjk)
{
    const int d   = threadIdx.x;
    const int idx = blockIdx.x;   // k for Pij; j for Pik; i for Pjk
    const int b   = blockIdx.y;
    const size_t pb = (size_t)b * NP;
    float sij = 0.f, sik = 0.f, sjk = 0.f;
    for (int q = 0; q < NN; ++q) {
        sij += Pi[(pb + q * NN + idx) * ND + d];
        sik += Pi[(pb + idx * NN + q) * ND + d];
        sjk += Pj[(pb + idx * NN + q) * ND + d];
    }
    const size_t ob = ((size_t)b * NN + idx) * ND + d;
    const float inv = 1.0f / (float)NN;
    Pij[ob] = sij * inv; Pik[ob] = sik * inv; Pjk[ob] = sjk * inv;
}

// Pa = mean_k Pij.  grid(64) x 128
__global__ __launch_bounds__(128) void pool3_kernel(
    const float* __restrict__ Pij, float* __restrict__ Pa)
{
    const int d = threadIdx.x;
    const int b = blockIdx.x;
    float s = 0.f;
    for (int q = 0; q < NN; ++q) s += Pij[((size_t)b * NN + q) * ND + d];
    Pa[(size_t)b * ND + d] = s * (1.0f / (float)NN);
}

// ---------------------------------------------------------------------------
// Corrections: Cjk = W1·Pi + W4·Pij + W5·Pik + W7·Pa + bias
//              Cik = W2·Pj + W6·Pjk ;  Cij = W3·Pk
// grid(169,64,3) x 128 (thread = output channel e)
// ---------------------------------------------------------------------------
__global__ __launch_bounds__(128) void corr_kernel(
    const float* __restrict__ Pi, const float* __restrict__ Pj,
    const float* __restrict__ Pk, const float* __restrict__ Pij,
    const float* __restrict__ Pik, const float* __restrict__ Pjk,
    const float* __restrict__ Pa,
    const float* __restrict__ eqW, const float* __restrict__ eqb,
    float* __restrict__ Cjk, float* __restrict__ Cik, float* __restrict__ Cij)
{
    __shared__ float pA[ND], pB[ND], pC[ND], pD[ND];
    const int e    = threadIdx.x;
    const int pair = blockIdx.x;
    const int b    = blockIdx.y;
    const int var  = blockIdx.z;
    const int hi   = pair / NN;
    const int lo   = pair - hi * NN;
    const size_t pbase = ((size_t)b * NP + pair) * ND;

    if (var == 0) {                      // pair = (j,k)
        pA[e] = Pi [pbase + e];
        pB[e] = Pij[((size_t)b * NN + lo) * ND + e];
        pC[e] = Pik[((size_t)b * NN + hi) * ND + e];
        pD[e] = Pa [(size_t)b * ND + e];
        __syncthreads();
        const float* W1 = eqW + (size_t)1 * ND * ND;
        const float* W4 = eqW + (size_t)4 * ND * ND;
        const float* W5 = eqW + (size_t)5 * ND * ND;
        const float* W7 = eqW + (size_t)7 * ND * ND;
        float o = eqb[e];
        for (int d = 0; d < ND; ++d)
            o += W1[d * ND + e] * pA[d] + W4[d * ND + e] * pB[d]
               + W5[d * ND + e] * pC[d] + W7[d * ND + e] * pD[d];
        Cjk[pbase + e] = o;
    } else if (var == 1) {               // pair = (i,k)
        pA[e] = Pj [pbase + e];
        pB[e] = Pjk[((size_t)b * NN + hi) * ND + e];
        __syncthreads();
        const float* W2 = eqW + (size_t)2 * ND * ND;
        const float* W6 = eqW + (size_t)6 * ND * ND;
        float o = 0.f;
        for (int d = 0; d < ND; ++d)
            o += W2[d * ND + e] * pA[d] + W6[d * ND + e] * pB[d];
        Cik[pbase + e] = o;
    } else {                             // pair = (i,j)
        pA[e] = Pk[pbase + e];
        __syncthreads();
        const float* W3 = eqW + (size_t)3 * ND * ND;
        float o = 0.f;
        for (int d = 0; d < ND; ++d) o += W3[d * ND + e] * pA[d];
        Cij[pbase + e] = o;
    }
}

// ---------------------------------------------------------------------------
// Main WMMA GEMM: out[b, m, e] = relu( sum_d T[b,m,d]*W0[d,e] + corrections ).
// M = 2197 rows/batch, K = N = 128. grid(18, 64) x 256 (8 waves).
// Each wave owns a 16-row x 128-col strip: 4 k-steps x 8 n-tiles of
// v_wmma_f32_16x16x32_bf16. W0 staged in LDS as bf16, transposed [e][k],
// row stride 136 (bank-conflict-free). A fragments: two 16B global loads.
// doStore: write bf16 tensor. doVsum: ReLU row-reduction -> atomic v[b,e].
// ---------------------------------------------------------------------------
__global__ __launch_bounds__(256) void eq_gemm_kernel(
    const __bf16* __restrict__ T, const float* __restrict__ W0,
    const float* __restrict__ Cjk, const float* __restrict__ Cik,
    const float* __restrict__ Cij,
    __bf16* __restrict__ Tout, float* __restrict__ vout,
    int doStore, int doVsum)
{
    __shared__ __bf16 Wlds[ND * 136];
    __shared__ float  colsum[ND];

    const int tid = threadIdx.x;
    const int rb  = blockIdx.x;     // row-block (128 rows) within batch element
    const int b   = blockIdx.y;

    // Stage W0 (f32 [k][e]) -> Wlds[e*136 + k] in bf16 (coalesced reads).
    for (int idx = tid; idx < ND * ND; idx += 256) {
        const int k = idx >> 7;
        const int e = idx & 127;
        Wlds[e * 136 + k] = (__bf16)W0[idx];
    }
    if (tid < ND) colsum[tid] = 0.0f;
    __syncthreads();

    const int wave    = tid >> 5;
    const int lane    = tid & 31;
    const int halfsel = lane >> 4;      // 0 | 1
    const int ln      = lane & 15;

    // A-fragment row for this lane (per ISA 16-bit A 16x32 layout).
    int aRow = rb * 128 + wave * 16 + ln;
    const int aRowC = aRow < NC ? aRow : (NC - 1);     // clamp; masked later
    const __bf16* Arow = T + ((size_t)b * NC + (size_t)aRowC) * ND;
    const int aoff = halfsel * 8;                      // K offset 0 or 8

    v8f acc[8];
#pragma unroll
    for (int t = 0; t < 8; ++t)
#pragma unroll
        for (int q = 0; q < 8; ++q) acc[t][q] = 0.0f;

#pragma unroll
    for (int ks = 0; ks < 4; ++ks) {
        // A: K runs [ks*32+aoff, +8) and [ks*32+16+aoff, +8) -> two 16B loads.
        const __bf16* p0 = Arow + ks * 32 + aoff;
        const __bf16* p1 = Arow + ks * 32 + 16 + aoff;
        const v16bf afrag = pack16(*(const v8bf*)p0, *(const v8bf*)p1);
        const int kb2 = ks * 32 + halfsel * 8;         // B: same K runs as A
#pragma unroll
        for (int t = 0; t < 8; ++t) {
            const __bf16* q = &Wlds[(t * 16 + ln) * 136 + kb2];
            const v16bf bfrag = pack16(*(const v8bf*)q, *(const v8bf*)(q + 16));
            acc[t] = __builtin_amdgcn_wmma_f32_16x16x32_bf16(
                false, afrag, false, bfrag, (short)0, acc[t], false, false);
        }
    }

    // Epilogue: C/D layout -> row = rowTop + r, col = t*16 + ln.
    const int rowTop = rb * 128 + wave * 16 + halfsel * 8;
#pragma unroll
    for (int t = 0; t < 8; ++t) {
        const int col = t * 16 + ln;
        float lsum = 0.0f;
#pragma unroll
        for (int r = 0; r < 8; ++r) {
            const int row = rowTop + r;
            if (row < NC) {
                const int i   = row / NP;
                const int rem = row - i * NP;
                const int j   = rem / NN;
                const int k   = rem - j * NN;
                float val = acc[t][r]
                    + Cjk[((size_t)b * NP + rem)           * ND + col]
                    + Cik[((size_t)b * NP + i * NN + k)    * ND + col]
                    + Cij[((size_t)b * NP + i * NN + j)    * ND + col];
                val = frelu(val);
                if (doStore)
                    Tout[((size_t)b * NC + row) * ND + col] = (__bf16)val;
                lsum += val;
            }
        }
        if (doVsum) atomicAdd(&colsum[col], lsum);
    }
    if (doVsum) {
        __syncthreads();
        if (tid < ND) atomicAdd(&vout[(size_t)b * ND + tid], colsum[tid]);
    }
}

// ---------------------------------------------------------------------------
__global__ __launch_bounds__(256) void zero_kernel(float* __restrict__ p, int n)
{
    const int i = blockIdx.x * 256 + threadIdx.x;
    if (i < n) p[i] = 0.0f;
}

// Decoder MLP: v[b,128] -> out[b,10].  grid(64) x 128
__global__ __launch_bounds__(128) void decoder_kernel(
    const float* __restrict__ v,
    const float* __restrict__ W1, const float* __restrict__ b1,
    const float* __restrict__ W2, const float* __restrict__ b2,
    const float* __restrict__ W3, const float* __restrict__ b3,
    float* __restrict__ out)
{
    __shared__ float vl[ND], h1[ND], h2[ND];
    const int e = threadIdx.x;
    const int b = blockIdx.x;
    vl[e] = v[(size_t)b * ND + e];
    __syncthreads();
    float a = b1[e];
    for (int d = 0; d < ND; ++d) a += vl[d] * W1[d * ND + e];
    h1[e] = frelu(a);
    __syncthreads();
    a = b2[e];
    for (int d = 0; d < ND; ++d) a += h1[d] * W2[d * ND + e];
    h2[e] = frelu(a);
    __syncthreads();
    if (e < NOUT) {
        float o = b3[e];
        for (int d = 0; d < ND; ++d) o += h2[d] * W3[d * NOUT + e];
        out[(size_t)b * NOUT + e] = o;
    }
}

// ---------------------------------------------------------------------------
extern "C" void kernel_launch(void* const* d_in, const int* in_sizes, int n_in,
                              void* d_out, int out_size, void* d_ws, size_t ws_size,
                              hipStream_t stream)
{
    (void)in_sizes; (void)n_in; (void)out_size; (void)ws_size;

    const int*   suits  = (const int*)  d_in[0];
    const int*   nums   = (const int*)  d_in[1];
    const float* encW1  = (const float*)d_in[2];
    const float* encb1  = (const float*)d_in[3];
    const float* encW2  = (const float*)d_in[4];
    const float* encb2  = (const float*)d_in[5];
    const float* encW3  = (const float*)d_in[6];
    const float* encb3  = (const float*)d_in[7];
    const float* eq1W   = (const float*)d_in[8];
    const float* eq1b   = (const float*)d_in[9];
    const float* eq2W   = (const float*)d_in[10];
    const float* eq2b   = (const float*)d_in[11];
    const float* decW1  = (const float*)d_in[12];
    const float* decb1  = (const float*)d_in[13];
    const float* decW2  = (const float*)d_in[14];
    const float* decb2  = (const float*)d_in[15];
    const float* decW3  = (const float*)d_in[16];
    const float* decb3  = (const float*)d_in[17];
    float* out = (float*)d_out;

    // Workspace layout (all sizes already 256B multiples).
    const size_t X_BYTES  = (size_t)NB * NN * ND * 4;   // x fp32
    const size_t T_BYTES  = (size_t)NB * NC * ND * 2;   // tensor bf16
    const size_t PP_BYTES = (size_t)NB * NP * ND * 4;   // pair pools / corrections
    const size_t PL_BYTES = (size_t)NB * NN * ND * 4;   // line pools
    const size_t PA_BYTES = (size_t)NB * ND * 4;

    char* w = (char*)d_ws;
    size_t off = 0;
    float*  x    = (float*) (w + off); off += X_BYTES;
    __bf16* T0   = (__bf16*)(w + off); off += T_BYTES;
    __bf16* T1   = (__bf16*)(w + off); off += T_BYTES;
    float*  Pi   = (float*) (w + off); off += PP_BYTES;
    float*  Pj   = (float*) (w + off); off += PP_BYTES;
    float*  Pk   = (float*) (w + off); off += PP_BYTES;
    float*  Pij  = (float*) (w + off); off += PL_BYTES;
    float*  Pik  = (float*) (w + off); off += PL_BYTES;
    float*  Pjk  = (float*) (w + off); off += PL_BYTES;
    float*  Pa   = (float*) (w + off); off += PA_BYTES;
    float*  Cjk  = (float*) (w + off); off += PP_BYTES;
    float*  Cik  = (float*) (w + off); off += PP_BYTES;
    float*  Cij  = (float*) (w + off); off += PP_BYTES;
    float*  vbuf = (float*) (w + off); off += PA_BYTES;

    const dim3 gEnc(NN, NB);
    const dim3 gP1(NP, NB);
    const dim3 gP2(NN, NB);
    const dim3 gCorr(NP, NB, 3);
    const dim3 gGemm((NC + 127) / 128, NB);   // 18 x 64

    // Encoder + order-3 tensor (bf16).
    encoder_kernel<<<gEnc, 128, 0, stream>>>(suits, nums, encW1, encb1,
                                             encW2, encb2, encW3, encb3, x);
    outer3_kernel<<<gEnc, 256, 0, stream>>>(x, T0);

    // eq layer 1: pools -> corrections -> WMMA GEMM (store T1 bf16).
    pool1_kernel<<<gP1, 128, 0, stream>>>(T0, Pi, Pj, Pk);
    pool2_kernel<<<gP2, 128, 0, stream>>>(Pi, Pj, Pij, Pik, Pjk);
    pool3_kernel<<<NB, 128, 0, stream>>>(Pij, Pa);
    corr_kernel<<<gCorr, 128, 0, stream>>>(Pi, Pj, Pk, Pij, Pik, Pjk, Pa,
                                           eq1W, eq1b, Cjk, Cik, Cij);
    eq_gemm_kernel<<<gGemm, 256, 0, stream>>>(T0, eq1W, Cjk, Cik, Cij,
                                              T1, vbuf, /*store=*/1, /*vsum=*/0);

    // eq layer 2: pools -> corrections -> WMMA GEMM (fused ReLU + sum -> v).
    pool1_kernel<<<gP1, 128, 0, stream>>>(T1, Pi, Pj, Pk);
    pool2_kernel<<<gP2, 128, 0, stream>>>(Pi, Pj, Pij, Pik, Pjk);
    pool3_kernel<<<NB, 128, 0, stream>>>(Pij, Pa);
    corr_kernel<<<gCorr, 128, 0, stream>>>(Pi, Pj, Pk, Pij, Pik, Pjk, Pa,
                                           eq2W, eq2b, Cjk, Cik, Cij);
    zero_kernel<<<(NB * ND + 255) / 256, 256, 0, stream>>>(vbuf, NB * ND);
    eq_gemm_kernel<<<gGemm, 256, 0, stream>>>(T1, eq2W, Cjk, Cik, Cij,
                                              T0 /*unused*/, vbuf,
                                              /*store=*/0, /*vsum=*/1);

    // Decoder.
    decoder_kernel<<<NB, 128, 0, stream>>>(vbuf, decW1, decb1, decW2, decb2,
                                           decW3, decb3, out);
}
